// CustomGATEncoder_64141041598738
// MI455X (gfx1250) — matrix operand order repaired and checked by reading, not compiled, not run on hardware
//
#include <hip/hip_runtime.h>
#include <hip/hip_bf16.h>
#include <math.h>

typedef __attribute__((ext_vector_type(16))) _Float16 v16h;
typedef __attribute__((ext_vector_type(8)))  _Float16 v8h;
typedef __attribute__((ext_vector_type(8)))  float    v8f;

static constexpr int Bn = 64;
static constexpr int Nn = 1000;
static constexpr int Np = 1024;          // padded to 64 tiles of 16
static constexpr int NODES = Bn * Nn;    // 64000
static constexpr float NEG = 0.2f;
static constexpr float EPS = 1e-5f;

// ---------------------------------------------------------------------------
// Location embedding MLP: [2]->64 ReLU ->64 BN ReLU ->128, one node per block
// ---------------------------------------------------------------------------
__global__ void mlp_kernel(const float* __restrict__ locs,
                           const float* __restrict__ W0, const float* __restrict__ b0,
                           const float* __restrict__ W1, const float* __restrict__ b1,
                           const float* __restrict__ g,  const float* __restrict__ bb,
                           const float* __restrict__ m,  const float* __restrict__ v,
                           const float* __restrict__ Wo, const float* __restrict__ bo,
                           float* __restrict__ xout)
{
    __shared__ float z0[64], z1[64];
    const int node = blockIdx.x, t = threadIdx.x;
    const float lx = locs[node * 2 + 0], ly = locs[node * 2 + 1];
    if (t < 64) {
        float s = lx * W0[t] + ly * W0[64 + t] + b0[t];
        z0[t] = s > 0.f ? s : 0.f;
    }
    __syncthreads();
    if (t < 64) {
        float s = b1[t];
#pragma unroll 8
        for (int k = 0; k < 64; ++k) s += z0[k] * W1[k * 64 + t];
        s = (s - m[t]) * rsqrtf(v[t] + EPS) * g[t] + bb[t];
        z1[t] = s > 0.f ? s : 0.f;
    }
    __syncthreads();
    float s = bo[t];
#pragma unroll 8
    for (int k = 0; k < 64; ++k) s += z1[k] * Wo[k * 128 + t];
    xout[(size_t)node * 128 + t] = s;
}

// ---------------------------------------------------------------------------
// Convert fp32 GAT weights to f16, transposed: w16t[l][n][k] = W[l][k][n]
// ---------------------------------------------------------------------------
__global__ void wconvT_kernel(const float* __restrict__ w, _Float16* __restrict__ w16t)
{
    const int idx = blockIdx.x * 256 + threadIdx.x;
    if (idx < 3 * 128 * 128) {
        const int l = idx >> 14;
        const int n = (idx >> 7) & 127;
        const int k = idx & 127;
        w16t[idx] = (_Float16)w[(l << 14) + k * 128 + n];
    }
}

// ---------------------------------------------------------------------------
// h = x @ W via WMMA. Grid (64 j-tiles, 64 batches), 256 threads = 8 waves,
// wave w owns n-columns [16w,16w+16). Epilogue stores the 16x16 D tile as a
// single packed b128 per lane directly into the TRANSPOSED h16T[b][n][j]
// layout (lane holds 8 consecutive j for one n), zero-padding j>=1000.
// ---------------------------------------------------------------------------
__global__ void gemm_kernel(const float* __restrict__ x,       // [B*N,128] f32
                            const _Float16* __restrict__ w16t, // [128 n][128 k] f16
                            _Float16* __restrict__ h16t)       // [B][128][1024] f16
{
    constexpr int LDW = 136;
    __shared__ __align__(16) _Float16 WshT[128 * LDW]; // [n][k]
    __shared__ __align__(16) _Float16 Xsh[16 * LDW];   // [m][k]
    const int t  = threadIdx.x;
    const int ti = blockIdx.x;           // 0..63
    const int b  = blockIdx.y;
    const int j0 = ti * 16;

    const unsigned* wsrc = (const unsigned*)w16t;
    unsigned* wdst = (unsigned*)WshT;
    for (int lin = t; lin < 8192; lin += 256) {
        const int r = lin >> 6, c = lin & 63;
        wdst[r * (LDW / 2) + c] = wsrc[lin];
    }
    for (int lin = t; lin < 2048; lin += 256) {
        const int r = lin >> 7, c = lin & 127;
        const int node = j0 + r;
        Xsh[r * LDW + c] = (node < Nn)
            ? (_Float16)x[((size_t)b * Nn + node) * 128 + c] : (_Float16)0.f;
    }
    __syncthreads();

    const int wave = t >> 5, lane = t & 31;
    const int halfL = lane >> 4, n0 = lane & 15;
    const int cb = wave * 16, arow = lane & 15;

    v8f acc = {};
#pragma unroll
    for (int kk = 0; kk < 4; ++kk) {
        v16h a, bm;
        const int ka = kk * 32 + halfL * 8;
#pragma unroll
        for (int q = 0; q < 8; ++q) {
            a[q]     = Xsh[arow * LDW + ka + q];
            a[8 + q] = Xsh[arow * LDW + ka + 16 + q];
        }
        const int kb = kk * 32 + halfL * 16;
#pragma unroll
        for (int q = 0; q < 16; ++q)
            bm[q] = WshT[(cb + n0) * LDW + kk * 32 + halfL * 8 + (q & 7) + ((q >> 3) << 4)];
        acc = __builtin_amdgcn_wmma_f32_16x16x32_f16(false, a, false, bm,
                                                     (short)0, acc, false, false);
    }
    v8h pk;
#pragma unroll
    for (int vv = 0; vv < 8; ++vv) pk[vv] = (_Float16)acc[vv];
    *(v8h*)&h16t[((size_t)b * 128 + cb + n0) * Np + j0 + halfL * 8] = pk;
}

// ---------------------------------------------------------------------------
// e_src[b,j] = h[b,j,:].a_src ; e_dst likewise. Coalesced over j via h16T.
// ---------------------------------------------------------------------------
__global__ void edot_kernel(const _Float16* __restrict__ h16t,
                            const float* __restrict__ asrc, const float* __restrict__ adst,
                            float* __restrict__ es, float* __restrict__ ed)
{
    const int j = blockIdx.x * 256 + threadIdx.x;
    const int b = blockIdx.y;
    if (j >= Nn) return;
    float s1 = 0.f, s2 = 0.f;
#pragma unroll 8
    for (int d = 0; d < 128; ++d) {
        const float hv = (float)h16t[((size_t)b * 128 + d) * Np + j];
        s1 += hv * asrc[d];
        s2 += hv * adst[d];
    }
    es[b * Nn + j] = s1;
    ed[b * Nn + j] = s2;
}

// ---------------------------------------------------------------------------
// Fused dense-GAT attention. Grid (16 blocks of 64 target rows, 64 batches),
// 256 threads = 8 waves (wave = n-tile). Each iteration stages a 64-wide
// source slab (K=64) and issues 8 WMMAs per wave (4 i-tiles x 2 k-halves)
// against it. Exact softmax max via monotone leaky-relu of separable logits.
// ---------------------------------------------------------------------------
__global__ void attn_kernel(const _Float16* __restrict__ h16t,
                            const float* __restrict__ e_src, const float* __restrict__ e_dst,
                            const float* __restrict__ bias,
                            const float* __restrict__ bng, const float* __restrict__ bnb,
                            const float* __restrict__ bnm, const float* __restrict__ bnv,
                            int apply_bn, float* __restrict__ out)
{
    constexpr int LDT = 72;   // halves per HsT/Ps row (16B-aligned, padded)
    __shared__ float es_sh[1024];
    __shared__ float red[256];
    __shared__ float edsh[64], msh[64], denom[64];
    __shared__ __align__(16) _Float16 HsT[128 * LDT]; // [n][k]   (k = 64 slab)
    __shared__ __align__(16) _Float16 Ps[64 * LDT];   // [i][k]

    const int t  = threadIdx.x;
    const int bi = blockIdx.x;       // block of 64 target rows (0..15)
    const int b  = blockIdx.y;       // batch
    const int i0 = bi * 64;

    for (int lin = t; lin < 1024; lin += 256)
        es_sh[lin] = (lin < Nn) ? e_src[b * Nn + lin] : -3.0e38f;

    float lm = -3.0e38f;
    for (int lin = t; lin < Nn; lin += 256)
        lm = fmaxf(lm, e_src[b * Nn + lin]);
    red[t] = lm;
    __syncthreads();
    for (int s = 128; s > 0; s >>= 1) {
        if (t < s) red[t] = fmaxf(red[t], red[t + s]);
        __syncthreads();
    }
    const float smax = red[0];
    if (t < 64) {
        const int i = i0 + t;
        const float ed = (i < Nn) ? e_dst[b * Nn + i] : 0.f;
        edsh[t] = ed;
        const float mm = smax + ed;
        msh[t] = mm > 0.f ? mm : NEG * mm;   // exact row max of lrelu logits
        denom[t] = 0.f;
    }
    __syncthreads();

    const int wave = t >> 5, lane = t & 31;
    const int halfL = lane >> 4, n0 = lane & 15;
    const int cb = wave * 16, arow = lane & 15;
    v8f acc0 = {}, acc1 = {}, acc2 = {}, acc3 = {};
    const unsigned* hsrc = (const unsigned*)h16t;   // row stride Np/2 dwords
    unsigned* hdst = (unsigned*)HsT;

    for (int jb = 0; jb < 16; ++jb) {
        const int j0 = jb * 64;
        // stage HsT[n][0..63] <- h16T[b][n][j0..j0+63]  (128 rows x 32 dwords)
        for (int lin = t; lin < 4096; lin += 256) {
            const int n = lin >> 5, c = lin & 31;
            hdst[n * (LDT / 2) + c] =
                hsrc[((size_t)b * 128 + n) * (Np / 2) + (j0 >> 1) + c];
        }
        // P[i,jj] = exp(lrelu(e_src[j]+e_dst[i]) - m_i)  (f16 in LDS)
        for (int lin = t; lin < 4096; lin += 256) {
            const int i = lin >> 6, jj = lin & 63;
            const float e  = es_sh[j0 + jj] + edsh[i];
            const float lr = e > 0.f ? e : NEG * e;
            Ps[i * LDT + jj] = (_Float16)__expf(lr - msh[i]);
        }
        // prefetch next source slab (probe-confirmed -> global_prefetch_b8)
        if (jb + 1 < 16 && t < 128) {
            __builtin_prefetch(&h16t[((size_t)b * 128 + t) * Np + j0 + 64], 0, 1);
        }
        __syncthreads();
        if (t < 64) {
            float s = denom[t];
#pragma unroll 8
            for (int jj = 0; jj < 64; ++jj) s += (float)Ps[t * LDT + jj];
            denom[t] = s;
        }
#pragma unroll
        for (int kk = 0; kk < 2; ++kk) {
            v16h bm;
#pragma unroll
            for (int q = 0; q < 16; ++q)
                bm[q] = HsT[(cb + n0) * LDT + kk * 32 + halfL * 8 + (q & 7) + ((q >> 3) << 4)];
#pragma unroll
            for (int it = 0; it < 4; ++it) {
                v16h a;
                const int rr = it * 16 + arow;
                const int ka = kk * 32 + halfL * 8;
#pragma unroll
                for (int q = 0; q < 8; ++q) {
                    a[q]     = Ps[rr * LDT + ka + q];
                    a[8 + q] = Ps[rr * LDT + ka + 16 + q];
                }
                v8f& acc = (it == 0) ? acc0 : (it == 1) ? acc1 : (it == 2) ? acc2 : acc3;
                acc = __builtin_amdgcn_wmma_f32_16x16x32_f16(false, a, false, bm,
                                                             (short)0, acc, false, false);
            }
        }
        __syncthreads();
    }

#pragma unroll
    for (int it = 0; it < 4; ++it) {
        const v8f& acc = (it == 0) ? acc0 : (it == 1) ? acc1 : (it == 2) ? acc2 : acc3;
#pragma unroll
        for (int vv = 0; vv < 8; ++vv) {
            const int mrow = it * 16 + vv + halfL * 8;
            const int i = i0 + mrow;
            if (i < Nn) {
                const int n = cb + n0;
                float val = acc[vv] / denom[mrow] + bias[n];
                if (apply_bn) {
                    val = (val - bnm[n]) * rsqrtf(bnv[n] + EPS) * bng[n] + bnb[n];
                    val = val > 0.f ? val : 0.f;
                }
                out[(size_t)(b * Nn + i) * 128 + n] = val;
            }
        }
    }
}

// ---------------------------------------------------------------------------
extern "C" void kernel_launch(void* const* d_in, const int* in_sizes, int n_in,
                              void* d_out, int out_size, void* d_ws, size_t ws_size,
                              hipStream_t stream)
{
    const float* locs    = (const float*)d_in[0];
    const float* le_W0   = (const float*)d_in[1];
    const float* le_b0   = (const float*)d_in[2];
    const float* le_W1   = (const float*)d_in[3];
    const float* le_b1   = (const float*)d_in[4];
    const float* le_bn_g = (const float*)d_in[5];
    const float* le_bn_b = (const float*)d_in[6];
    const float* le_bn_m = (const float*)d_in[7];
    const float* le_bn_v = (const float*)d_in[8];
    const float* le_Wo   = (const float*)d_in[9];
    const float* le_bo   = (const float*)d_in[10];
    const float* gat_W   = (const float*)d_in[11];
    const float* gat_as  = (const float*)d_in[12];
    const float* gat_ad  = (const float*)d_in[13];
    const float* gat_b   = (const float*)d_in[14];
    const float* bn_g    = (const float*)d_in[15];
    const float* bn_b    = (const float*)d_in[16];
    const float* bn_m    = (const float*)d_in[17];
    const float* bn_v    = (const float*)d_in[18];

    char* ws = (char*)d_ws;
    size_t off = 0;
    auto alloc = [&](size_t bytes) -> void* {
        void* p = ws + off;
        off += (bytes + 255) & ~(size_t)255;
        return p;
    };
    float*    x0    = (float*)   alloc((size_t)NODES * 128 * 4);
    float*    x1    = (float*)   alloc((size_t)NODES * 128 * 4);
    _Float16* h16t  = (_Float16*)alloc((size_t)Bn * 128 * Np * 2);
    _Float16* w16t  = (_Float16*)alloc((size_t)3 * 128 * 128 * 2);
    float*    esrc  = (float*)   alloc((size_t)NODES * 4);
    float*    edst  = (float*)   alloc((size_t)NODES * 4);

    mlp_kernel<<<NODES, 128, 0, stream>>>(locs, le_W0, le_b0, le_W1, le_b1,
                                          le_bn_g, le_bn_b, le_bn_m, le_bn_v,
                                          le_Wo, le_bo, x0);
    wconvT_kernel<<<(3 * 128 * 128 + 255) / 256, 256, 0, stream>>>(gat_W, w16t);

    const float* xin = x0;
    for (int l = 0; l < 3; ++l) {
        gemm_kernel<<<dim3(64, Bn), 256, 0, stream>>>(xin, w16t + (size_t)l * 128 * 128, h16t);
        edot_kernel<<<dim3((Nn + 255) / 256, Bn), 256, 0, stream>>>(h16t, gat_as + l * 128,
                                                                    gat_ad + l * 128, esrc, edst);
        float* xout = (l == 2) ? (float*)d_out : ((l == 0) ? x1 : x0);
        const int bl = (l < 2) ? l : 0;
        attn_kernel<<<dim3(16, Bn), 256, 0, stream>>>(h16t, esrc, edst, gat_b + l * 128,
                                                      bn_g + bl * 128, bn_b + bl * 128,
                                                      bn_m + bl * 128, bn_v + bl * 128,
                                                      (l < 2) ? 1 : 0, xout);
        xin = xout;
    }
}
